// DCT_1262720385591
// MI455X (gfx1250) — compile-verified, matrix operand order checked by reading, not run on hardware
//
#include <hip/hip_runtime.h>

typedef __attribute__((ext_vector_type(16))) _Float16 v16h;
typedef __attribute__((ext_vector_type(8)))  _Float16 v8h;
typedef __attribute__((ext_vector_type(8)))  float    v8f;
typedef __attribute__((ext_vector_type(4)))  float    v4f;

#define DIM 512
#define BK   32    // WMMA K per step
#define BLK 128    // block tile edge
#define BST  40    // img staging tile stride in halves (32 + 8 pad)
#define TST 520    // T' tile stride in halves (512 + 8 pad)
#define NI    4    // 16x16 tiles per wave along i (4*16 = 64)
#define NJ    2    // 16x16 tiles per wave along j (2*16 = 32)

// ---------------------------------------------------------------------------
// Cosine table: C[i][j] = cos(pi*(2j+1)*i/1024), f16.  Exact integer phase
// reduction mod 2048 keeps cosf's argument small and accurate.
// ---------------------------------------------------------------------------
__global__ __launch_bounds__(256) void dct_cos_table(_Float16* __restrict__ tab) {
    int idx = blockIdx.x * 256 + threadIdx.x;     // 1024 blocks * 256 = 512*512
    int i = idx >> 9;
    int j = idx & 511;
    int t = ((2 * j + 1) * i) & 2047;             // phase in units of pi/1024
    float v = __cosf((float)t * 3.0679615757712823e-3f);  // * pi/1024
    tab[idx] = (_Float16)v;
}

__device__ __forceinline__ v16h join16(v8h lo, v8h hi) {
    return __builtin_shufflevector(lo, hi, 0,1,2,3,4,5,6,7,8,9,10,11,12,13,14,15);
}

// Fragment load from LDS, row-major over K.  Lane L: row = base+(L&15);
// halves 0-7 at k = sel, halves 8-15 at k = 16+sel, sel = 8*(L>=16).
// (ISA 16-bit A-matrix 16x32 layout; B mirrors it as rows of B^T.)
__device__ __forceinline__ v16h lfrag(const _Float16* base, int row,
                                      int stride, int kbase, int lane) {
    const _Float16* p = base + (row + (lane & 15)) * stride + kbase + ((lane >> 4) << 3);
    v8h lo = *(const v8h*)(p);        // ds_load_b128
    v8h hi = *(const v8h*)(p + 16);   // ds_load_b128
    return join16(lo, hi);
}

// Fragment load straight from the (L2-resident) global cosine table.
// rowbase already includes (row + (lane&15))*DIM + sel.
__device__ __forceinline__ v16h gfrag(const _Float16* rowbase, int kk) {
    v8h lo = *(const v8h*)(rowbase + kk);        // global_load_b128 (L2 hit)
    v8h hi = *(const v8h*)(rowbase + kk + 16);   // global_load_b128 (L2 hit)
    return join16(lo, hi);
}

// Load 16 img floats and convert to two v8h (for LDS staging).
__device__ __forceinline__ void loadcvt16(const float* src, v8h* h0, v8h* h1) {
    const v4f* s4 = (const v4f*)src;
    v4f f0 = s4[0], f1 = s4[1], f2 = s4[2], f3 = s4[3];
    #pragma unroll
    for (int e = 0; e < 4; ++e) {
        (*h0)[e]     = (_Float16)f0[e];
        (*h0)[e + 4] = (_Float16)f1[e];
        (*h1)[e]     = (_Float16)f2[e];
        (*h1)[e + 4] = (_Float16)f3[e];
    }
}

// ---------------------------------------------------------------------------
// Fused separable DCT for one (batch, q-block):
//   GEMM1:  T'[q][m] = sum_n C[q][n] * X[m][n]   (T' kept resident in LDS, f16)
//   GEMM2:  Y[p][q]  = scale * sum_m C[p][m] * T'[q][m]   (barrier-free loop)
// ---------------------------------------------------------------------------
__global__ __launch_bounds__(256) void dct_fused(const float* __restrict__ img,
                                                 const _Float16* __restrict__ tab,
                                                 float* __restrict__ out) {
    extern __shared__ _Float16 smem[];
    _Float16* Tlds = smem;                        // [128][TST]    133,120 B
    _Float16* Bt0  = smem + BLK * TST;            // [128][BST]     10,240 B
    _Float16* Bt1  = Bt0 + BLK * BST;             // [128][BST]     10,240 B

    const int qb   = blockIdx.x;                  // 0..3   (q block)
    const int bat  = blockIdx.y;                  // 0..127 (batch)
    const int tid  = threadIdx.x;
    const int lane = tid & 31;
    const int wave = tid >> 5;                    // 8 wave32s
    const int wi   = (wave & 1) * (NI * 16);      // wave offset along i (0/64)
    const int wj   = (wave >> 1) * (NJ * 16);     // wave offset along j (0..96)
    const int lr   = lane & 15;
    const int sel  = (lane >> 4) << 3;            // 0 or 8
    const int hrow = lane >> 4;                   // C/D row-half select

    const int srow = tid >> 1;                    // staging row 0..127
    const int scol = (tid & 1) << 4;              // staging col 0 or 16

    // Per-fragment global row bases into the cosine table (GEMM1: q rows).
    const _Float16* aQ[NI];
    #pragma unroll
    for (int ti = 0; ti < NI; ++ti)
        aQ[ti] = tab + (size_t)(qb * BLK + wi + ti * 16 + lr) * DIM + sel;

    const float* imgB = img + (size_t)bat * DIM * DIM;

    // =============== GEMM1: build T'[q][m] in LDS (double-buffered B) ========
    for (int mblk = 0; mblk < 4; ++mblk) {
        v8f acc[NI][NJ] = {};
        const float* xrow = imgB + (size_t)(mblk * BLK + srow) * DIM + scol;

        // Prologue: stage img tile kk=0 into Bt0.
        v8h h0, h1;
        loadcvt16(xrow, &h0, &h1);
        *(v8h*)(Bt0 + srow * BST + scol)     = h0;
        *(v8h*)(Bt0 + srow * BST + scol + 8) = h1;
        __syncthreads();

        for (int kk = 0; kk < DIM; kk += BK) {
            _Float16* cur = (kk & BK) ? Bt1 : Bt0;
            _Float16* nxt = (kk & BK) ? Bt0 : Bt1;
            const bool more = (kk + BK) < DIM;

            // Prefetch next img tile into VGPRs while WMMAs run on `cur`.
            if (more) loadcvt16(xrow + kk + BK, &h0, &h1);

            v16h afr[NI], bfr[NJ];
            #pragma unroll
            for (int ti = 0; ti < NI; ++ti)
                afr[ti] = gfrag(aQ[ti], kk);               // cos table from L2
            #pragma unroll
            for (int tj = 0; tj < NJ; ++tj)
                bfr[tj] = lfrag(cur, wj + tj * 16, BST, 0, lane);
            #pragma unroll
            for (int ti = 0; ti < NI; ++ti)
                #pragma unroll
                for (int tj = 0; tj < NJ; ++tj)
                    acc[ti][tj] = __builtin_amdgcn_wmma_f32_16x16x32_f16(
                        false, afr[ti], false, bfr[tj],
                        (short)0, acc[ti][tj], false, false);

            if (more) {
                *(v8h*)(nxt + srow * BST + scol)     = h0;
                *(v8h*)(nxt + srow * BST + scol + 8) = h1;
            }
            __syncthreads();      // one barrier per k-step
        }

        // Scatter f32 accum -> f16 T' tile (row = q_local, col = m).
        #pragma unroll
        for (int ti = 0; ti < NI; ++ti) {
            #pragma unroll
            for (int tj = 0; tj < NJ; ++tj) {
                int iq = wi + ti * 16 + hrow * 8;
                int jm = mblk * BLK + wj + tj * 16 + lr;
                #pragma unroll
                for (int r = 0; r < 8; ++r)
                    Tlds[(iq + r) * TST + jm] = (_Float16)acc[ti][tj][r];
            }
        }
    }
    __syncthreads();   // T' complete & visible to all waves

    // =============== GEMM2: Y = Cp x T'  (no LDS writes -> no barriers) ======
    const float rs2 = 0.70710678118654752f;       // 1/sqrt(2)
    for (int pblk = 0; pblk < 4; ++pblk) {
        const _Float16* aP[NI];
        #pragma unroll
        for (int ti = 0; ti < NI; ++ti)
            aP[ti] = tab + (size_t)(pblk * BLK + wi + ti * 16 + lr) * DIM + sel;

        v8f acc[NI][NJ] = {};
        #pragma unroll 2
        for (int kk = 0; kk < DIM; kk += BK) {
            v16h afr[NI], bfr[NJ];
            #pragma unroll
            for (int ti = 0; ti < NI; ++ti)
                afr[ti] = gfrag(aP[ti], kk);               // cos table from L2
            #pragma unroll
            for (int tj = 0; tj < NJ; ++tj)                // B direct from T'
                bfr[tj] = lfrag(Tlds, wj + tj * 16, TST, kk, lane);
            #pragma unroll
            for (int ti = 0; ti < NI; ++ti)
                #pragma unroll
                for (int tj = 0; tj < NJ; ++tj)
                    acc[ti][tj] = __builtin_amdgcn_wmma_f32_16x16x32_f16(
                        false, afr[ti], false, bfr[tj],
                        (short)0, acc[ti][tj], false, false);
        }

        // Fused scale + store to global f32 output.
        #pragma unroll
        for (int ti = 0; ti < NI; ++ti) {
            #pragma unroll
            for (int tj = 0; tj < NJ; ++tj) {
                int pbase = pblk * BLK + wi + ti * 16 + hrow * 8;
                int qg    = qb * BLK + wj + tj * 16 + lr;
                float cs  = (qg == 0) ? (0.00390625f * rs2) : 0.00390625f;  // 2/512
                #pragma unroll
                for (int r = 0; r < 8; ++r) {
                    int p   = pbase + r;
                    float s = (p == 0) ? (cs * rs2) : cs;
                    out[((size_t)bat * DIM + p) * DIM + qg] = acc[ti][tj][r] * s;
                }
            }
        }
    }
}

extern "C" void kernel_launch(void* const* d_in, const int* in_sizes, int n_in,
                              void* d_out, int out_size, void* d_ws, size_t ws_size,
                              hipStream_t stream) {
    const float* img = (const float*)d_in[0];
    float*       out = (float*)d_out;
    _Float16*    tab = (_Float16*)d_ws;          // 512*512*2 = 512 KB of scratch

    // Build shared cosine table (Cp == Cq since M == N == 512).
    dct_cos_table<<<dim3(1024), dim3(256), 0, stream>>>(tab);

    // Fused per-(batch, q-block) double-GEMM; T' lives in LDS (CDNA5: 320KB/WGP).
    size_t shmem = (size_t)(BLK * TST + 2 * BLK * BST) * sizeof(_Float16);  // 153,600 B
    dct_fused<<<dim3(4, 128), dim3(256), shmem, stream>>>(img, tab, out);
}
